// LocalitySelfAttention_67826123539067
// MI455X (gfx1250) — matrix operand order repaired and checked by reading, not compiled, not run on hardware
//
#include <hip/hip_runtime.h>

typedef __attribute__((ext_vector_type(16))) _Float16 v16h;
typedef __attribute__((ext_vector_type(8)))  _Float16 v8h;
typedef __attribute__((ext_vector_type(8)))  float    v8f;

typedef unsigned int u32;
typedef __attribute__((ext_vector_type(4))) u32 u32x4;
typedef __attribute__((ext_vector_type(8))) int i32x8;
typedef __attribute__((ext_vector_type(4))) int i32x4;

#define BB 2
#define LL 2048
#define HH 8
#define DD 64
#define BN 32               // keys per chunk
#define NCHUNK (LL / BN)
#define NWAVES 4
#define BM (16 * NWAVES)    // 64 query rows per workgroup
#define KROW 68             // staged f32 row stride (64 + 4 TDM pad dwords) = 272B
#define VSTR 40             // sVT f16 row stride (80B, 16B-aligned)

// Issue one 2D-tile TDM load: 32 rows x 64 f32, row stride 512 elems, with
// LDS padding of 4 dwords after every 64 dwords (row stride 272B in LDS).
__device__ __forceinline__ void tdm_load_2d(u32 lds_off, unsigned long long gaddr,
                                            const i32x8& g1) {
    u32x4 g0;
    g0.x = 1u;                                   // count=1 (valid user descriptor)
    g0.y = lds_off;                              // lds_addr (bytes)
    g0.z = (u32)gaddr;                           // global_addr[31:0]
    g0.w = (u32)(gaddr >> 32) | (2u << 30);      // global_addr[56:32] | type=2
    i32x4 z4 = {0, 0, 0, 0};                     // groups 2/3 unused (2D tensor)
    i32x8 z8 = {0, 0, 0, 0, 0, 0, 0, 0};
    __builtin_amdgcn_tensor_load_to_lds(g0, g1, z4, z4, z8, 0);
}

// Build interleaved A-operand fragment from f32 row: elems 0-7 = p[0..7],
// elems 8-15 = p[16..23]  (caller points p at row + 32*ds + 8*hig)
__device__ __forceinline__ v16h frag_from_f32(const float* p) {
    float4 a0 = *(const float4*)(p);
    float4 a1 = *(const float4*)(p + 4);
    float4 a2 = *(const float4*)(p + 16);
    float4 a3 = *(const float4*)(p + 20);
    v16h r;
    r[0]=(_Float16)a0.x; r[1]=(_Float16)a0.y; r[2]=(_Float16)a0.z; r[3]=(_Float16)a0.w;
    r[4]=(_Float16)a1.x; r[5]=(_Float16)a1.y; r[6]=(_Float16)a1.z; r[7]=(_Float16)a1.w;
    r[8]=(_Float16)a2.x; r[9]=(_Float16)a2.y; r[10]=(_Float16)a2.z; r[11]=(_Float16)a2.w;
    r[12]=(_Float16)a3.x; r[13]=(_Float16)a3.y; r[14]=(_Float16)a3.z; r[15]=(_Float16)a3.w;
    return r;
}

__global__ __launch_bounds__(128) void LocalitySelfAttention_fa_kernel(
    const float* __restrict__ Q, const float* __restrict__ K,
    const float* __restrict__ V, const float* __restrict__ scale,
    float* __restrict__ O)
{
    __shared__ float    stgK[2][BN][KROW];   // TDM-staged K chunk (f32, padded rows)
    __shared__ float    stgV[2][BN][KROW];   // TDM-staged V chunk (f32, padded rows)
    __shared__ _Float16 sVT[DD][VSTR];       // V chunk transposed+converted [d][key]

    const int tid  = threadIdx.x;
    const int wave = tid >> 5;
    const int lane = tid & 31;
    const int m    = lane & 15;   // query column owned by this lane
    const int hig  = lane >> 4;   // lane half (0/1)

    const int bh = blockIdx.y;
    const int b  = bh >> 3;       // / HH
    const int h  = bh & 7;        // % HH
    const int qbase = blockIdx.x * BM;

    // base-2 softmax: fold exp(scale) and log2(e) into one scalar
    const float factor2 = __expf(scale[0]) * 1.44269504088896340736f;

    // ---- TDM descriptor group 1 (constant): data_size=4B, pad 4dw per 64dw,
    //      tensor_dim0=64, tensor_dim1=2048, tile 64x32, stride0=512 elems
    i32x8 g1;
    g1[0] = (int)((2u << 16) | (1u << 20) | (5u << 22) | (3u << 25));
    g1[1] = (int)(64u << 16);      // tensor_dim0 lo16 -> bits[31:16]
    g1[2] = (int)(2048u << 16);    // td0 hi16=0 | tensor_dim1 lo16
    g1[3] = (int)(64u << 16);      // td1 hi16=0 | tile_dim0=64
    g1[4] = 32;                    // tile_dim1=32 | tile_dim2=0
    g1[5] = 512;                   // tensor_dim0_stride lo32
    g1[6] = 0;                     // stride0 hi16 | stride1 lo16
    g1[7] = 0;                     // stride1 hi32

    const unsigned long long kbase =
        (unsigned long long)(uintptr_t)K + (unsigned long long)((b * LL) * HH + h) * DD * 4ull;
    const unsigned long long vbase =
        (unsigned long long)(uintptr_t)V + (unsigned long long)((b * LL) * HH + h) * DD * 4ull;
    const unsigned long long chunk_bytes = (unsigned long long)BN * HH * DD * 4ull;

    // ---- per-wave Q fragments (B-operand layout: frag[i] = Q[m][32*ds + 8*hig + i], frag[8+i] = Q[m][32*ds + 16 + 8*hig + i])
    const int    qrow = qbase + wave * 16 + m;
    const float* qp   = Q + ((size_t)(b * LL + qrow) * HH + h) * DD;
    v16h qf[2];
    #pragma unroll
    for (int ds = 0; ds < 2; ++ds) {
        const float4* p4 = (const float4*)(qp + 32 * ds + 8 * hig);
        #pragma unroll
        for (int q = 0; q < 4; ++q) {
            float4 f = p4[q + (q & 2)];
            qf[ds][4*q+0] = (_Float16)f.x;
            qf[ds][4*q+1] = (_Float16)f.y;
            qf[ds][4*q+2] = (_Float16)f.z;
            qf[ds][4*q+3] = (_Float16)f.w;
        }
    }

    v8f  o[4] = {v8f{}, v8f{}, v8f{}, v8f{}};   // O^T accumulators (d tiles 0..3)
    float run_m = -3.0e38f;
    float run_l = 0.0f;

    // V transpose-pass indexing: 128 threads cover 32 rows x 64 floats
    const int ls = tid >> 2;      // key row within chunk (0..31)
    const int lq = tid & 3;       // 16-float quarter of the row

    // ---- kick off chunk 0 DMA
    if (wave == 0) {
        tdm_load_2d((u32)(uintptr_t)&stgK[0][0][0], kbase, g1);
        tdm_load_2d((u32)(uintptr_t)&stgV[0][0][0], vbase, g1);
    }

    for (int it = 0; it < NCHUNK; ++it) {
        const int buf = it & 1;
        if (wave == 0) __builtin_amdgcn_s_wait_tensorcnt(0);
        __syncthreads();   // stage[buf] arrived; all prior-chunk LDS reads done

        // prefetch next chunk into the other staging buffer (overlaps compute)
        if (wave == 0 && it + 1 < NCHUNK) {
            const unsigned long long off = (unsigned long long)(it + 1) * chunk_bytes;
            tdm_load_2d((u32)(uintptr_t)&stgK[buf ^ 1][0][0], kbase + off, g1);
            tdm_load_2d((u32)(uintptr_t)&stgV[buf ^ 1][0][0], vbase + off, g1);
        }

        // ---- cooperative V transpose + f16 convert: stgV[buf] -> sVT
        {
            const float* vrow = &stgV[buf][ls][lq * 16];
            #pragma unroll
            for (int q = 0; q < 4; ++q) {
                float4 vf = ((const float4*)vrow)[q];
                sVT[lq*16 + 4*q + 0][ls] = (_Float16)vf.x;
                sVT[lq*16 + 4*q + 1][ls] = (_Float16)vf.y;
                sVT[lq*16 + 4*q + 2][ls] = (_Float16)vf.z;
                sVT[lq*16 + 4*q + 3][ls] = (_Float16)vf.w;
            }
        }
        __syncthreads();

        // ---- S^T = K_tile * Q^T : two 16-key tiles, accumulate over d slices
        v8f st0 = {}, st1 = {};
        #pragma unroll
        for (int ds = 0; ds < 2; ++ds) {
            v16h ak0 = frag_from_f32(&stgK[buf][m     ][32*ds + 8*hig]);
            v16h ak1 = frag_from_f32(&stgK[buf][16 + m][32*ds + 8*hig]);
            st0 = __builtin_amdgcn_wmma_f32_16x16x32_f16(false, ak0, false, qf[ds], (short)0, st0, false, false);
            st1 = __builtin_amdgcn_wmma_f32_16x16x32_f16(false, ak1, false, qf[ds], (short)0, st1, false, false);
        }

        // ---- online softmax over this 32-key chunk (lane pair m <-> m+16)
        float sc0[8], sc1[8], mx = -3.0e38f;
        #pragma unroll
        for (int r = 0; r < 8; ++r) {
            sc0[r] = st0[r] * factor2;
            sc1[r] = st1[r] * factor2;
            mx = fmaxf(mx, fmaxf(sc0[r], sc1[r]));
        }
        mx = fmaxf(mx, __shfl_xor(mx, 16, 32));
        const float m_new = fmaxf(run_m, mx);
        const float cf    = __builtin_amdgcn_exp2f(run_m - m_new);
        float p0[8], p1[8], lsum = 0.0f;
        #pragma unroll
        for (int r = 0; r < 8; ++r) {
            p0[r] = __builtin_amdgcn_exp2f(sc0[r] - m_new);
            p1[r] = __builtin_amdgcn_exp2f(sc1[r] - m_new);
            lsum += p0[r] + p1[r];
        }
        lsum += __shfl_xor(lsum, 16, 32);
        run_l = run_l * cf + lsum;
        run_m = m_new;

        #pragma unroll
        for (int t = 0; t < 4; ++t)
            #pragma unroll
            for (int r = 0; r < 8; ++r) o[t][r] *= cf;

        // ---- P^T as B operand: B wants frag[i] = P_T[16*hig + i][m].
        // We hold P_T[r + 8*hig (+16)][m]; swap middle halves with lane^16.
        unsigned pk0[4], pk1[4];
        #pragma unroll
        for (int q = 0; q < 4; ++q) {
            union { _Float16 hh[2]; unsigned u; } a, c;
            a.hh[0] = (_Float16)p0[2*q]; a.hh[1] = (_Float16)p0[2*q+1]; pk0[q] = a.u;
            c.hh[0] = (_Float16)p1[2*q]; c.hh[1] = (_Float16)p1[2*q+1]; pk1[q] = c.u;
        }
        union { v16h v; unsigned u[8]; } pb;
        #pragma unroll
        for (int q = 0; q < 4; ++q) {
            const unsigned ox0 = (unsigned)__shfl_xor((int)pk0[q], 16, 32);
            const unsigned ox1 = (unsigned)__shfl_xor((int)pk1[q], 16, 32);
            pb.u[q]     = hig ? ox1    : pk0[q];
            pb.u[q + 4] = hig ? pk1[q] : ox0;
        }

        // ---- O^T += V^T * P^T (four 16-row d tiles, K = full 32-key chunk)
        #pragma unroll
        for (int t = 0; t < 4; ++t) {
            v16h av;
            const v8h lo = *(const v8h*)&sVT[16*t + m][16*hig];
            const v8h hh = *(const v8h*)&sVT[16*t + m][16*hig + 8];
            #pragma unroll
            for (int i = 0; i < 8; ++i) { av[i] = lo[i]; av[i+8] = hh[i]; }
            o[t] = __builtin_amdgcn_wmma_f32_16x16x32_f16(false, av, false, pb.v, (short)0, o[t], false, false);
        }
    }

    // ---- epilogue: O[qrow][d] = O^T[d][m] / l,  d = 16*t + 8*hig + r
    const float inv_l = 1.0f / run_l;
    float* op = O + ((size_t)(b * LL + qrow) * HH + h) * DD;
    #pragma unroll
    for (int t = 0; t < 4; ++t) {
        float4 lo4 = {o[t][0]*inv_l, o[t][1]*inv_l, o[t][2]*inv_l, o[t][3]*inv_l};
        float4 hi4 = {o[t][4]*inv_l, o[t][5]*inv_l, o[t][6]*inv_l, o[t][7]*inv_l};
        *(float4*)(op + 16*t + 8*hig)     = lo4;
        *(float4*)(op + 16*t + 8*hig + 4) = hi4;
    }
}

extern "C" void kernel_launch(void* const* d_in, const int* in_sizes, int n_in,
                              void* d_out, int out_size, void* d_ws, size_t ws_size,
                              hipStream_t stream) {
    const float* Q     = (const float*)d_in[0];
    const float* K     = (const float*)d_in[1];
    const float* V     = (const float*)d_in[2];
    const float* scale = (const float*)d_in[3];
    float*       O     = (float*)d_out;
    (void)in_sizes; (void)n_in; (void)out_size; (void)d_ws; (void)ws_size;

    dim3 grid(LL / BM, BB * HH);   // 32 query blocks x 16 (batch*head)
    dim3 block(32 * NWAVES);
    LocalitySelfAttention_fa_kernel<<<grid, block, 0, stream>>>(Q, K, V, scale, O);
}
